// MoEGate_73804718014883
// MI455X (gfx1250) — matrix operand order repaired and checked by reading, not compiled, not run on hardware
//
#include <hip/hip_runtime.h>
#include <hip/hip_bf16.h>
#include <math.h>

#define TT     16384     // BSZ*SEQ tokens
#define HH     2048      // hidden
#define EE     256       // experts
#define BSZn   4
#define SEQQ   4096
#define NGROUP 8
#define TOPKG  4
#define TOPK   8
#define SCALEF 2.5f
#define ALPHAF 0.001f

#define ROWS   64        // token rows per block
#define SECT   128       // K-section staged in LDS
#define NSECT  (HH / SECT)
#define LDA    (SECT + 8)

typedef __attribute__((ext_vector_type(16))) __bf16 v16bf;
typedef __attribute__((ext_vector_type(8)))  __bf16 v8bf;
typedef __attribute__((ext_vector_type(8)))  float  v8f;
typedef __attribute__((ext_vector_type(4)))  float  v4f;

// ---------------------------------------------------------------------------
// Kernel 1: scores = sigmoid(x @ W^T) via split-bf16 WMMA (3-pass hi/lo).
// block: 512 threads = 16 waves; wave owns a 16-expert strip, block covers
// 64 token rows x all 256 experts. A panel staged in LDS as hi/lo bf16;
// next section's global loads are issued before compute (register pipeline)
// so HBM traffic stays in flight underneath the WMMAs.
// ---------------------------------------------------------------------------
__device__ __forceinline__ void load_sect(const float* __restrict__ x,
                                          int t0, int ks, int tid, v4f* dst)
{
    #pragma unroll
    for (int i = 0; i < 4; ++i) {
        int v = tid + i * 512;        // linear float4 index in 64x128 panel
        int r = v >> 5;               // row  (32 float4 per row)
        int k = (v & 31) * 4;         // k offset within section
        dst[i] = *reinterpret_cast<const v4f*>(
            x + (size_t)(t0 + r) * HH + ks + k);
    }
}

__global__ __launch_bounds__(512) void gate_gemm_sigmoid(
    const float* __restrict__ x, const float* __restrict__ w,
    float* __restrict__ scores)
{
    __shared__ __bf16 As_hi[ROWS][LDA];
    __shared__ __bf16 As_lo[ROWS][LDA];

    const int tid  = threadIdx.x;
    const int lane = tid & 31;
    const int wave = tid >> 5;            // 0..15 -> expert tile
    const int t0   = blockIdx.x * ROWS;   // token row base
    const int n0   = wave * 16;           // expert col base
    const int half = lane >> 4;           // 0: lanes 0-15, 1: lanes 16-31
    const int l16  = lane & 15;

    v8f acc[4] = {};                      // 4 stacked 16x16 row tiles

    v4f pre[4];
    load_sect(x, t0, 0, tid, pre);        // prologue: section 0 in flight

    for (int s = 0; s < NSECT; ++s) {
        const int ks = s * SECT;
        __syncthreads();                  // LDS free (prev section consumed)
        // convert-once staging: f32 -> (hi,lo) bf16 planes in LDS
        #pragma unroll
        for (int i = 0; i < 4; ++i) {
            int v = tid + i * 512;
            int r = v >> 5;
            int k = (v & 31) * 4;
            #pragma unroll
            for (int q = 0; q < 4; ++q) {
                float f  = pre[i][q];
                __bf16 h = (__bf16)f;
                As_hi[r][k + q] = h;
                As_lo[r][k + q] = (__bf16)(f - (float)h);
            }
        }
        __syncthreads();
        if (s + 1 < NSECT)                // prefetch next section; latency
            load_sect(x, t0, ks + SECT, tid, pre);  // hidden by WMMAs below

        for (int k0 = 0; k0 < SECT; k0 += 32) {
            // ---- B fragment: B[k][n] = W[n][k]; lane holds 16 contiguous K
            const int kb = ks + k0 + half * 16;
            const float* wp = w + (size_t)(n0 + l16) * HH + kb;
            v16bf b_hi, b_lo;
            #pragma unroll
            for (int j = 0; j < 4; ++j) {
                v4f bv = *reinterpret_cast<const v4f*>(wp + j * 4);
                #pragma unroll
                for (int q = 0; q < 4; ++q) {
                    float f  = bv[q];
                    __bf16 h = (__bf16)f;
                    b_hi[j * 4 + q] = h;
                    b_lo[j * 4 + q] = (__bf16)(f - (float)h);
                }
            }
            // ---- A fragments per row-tile: two 8-element K runs per lane
            #pragma unroll
            for (int rt = 0; rt < 4; ++rt) {
                const int row = rt * 16 + l16;
                const int ka  = k0 + half * 16;
                v8bf h0 = *reinterpret_cast<const v8bf*>(&As_hi[row][ka]);
                v8bf h1 = *reinterpret_cast<const v8bf*>(&As_hi[row][ka + 8]);
                v8bf l0 = *reinterpret_cast<const v8bf*>(&As_lo[row][ka]);
                v8bf l1 = *reinterpret_cast<const v8bf*>(&As_lo[row][ka + 8]);
                v16bf a_hi, a_lo;
                #pragma unroll
                for (int q = 0; q < 8; ++q) {
                    a_hi[q] = h0[q]; a_hi[8 + q] = h1[q];
                    a_lo[q] = l0[q]; a_lo[8 + q] = l1[q];
                }
                acc[rt] = __builtin_amdgcn_wmma_f32_16x16x32_bf16(
                    false, a_hi, false, b_hi, (short)0, acc[rt], false, false);
                acc[rt] = __builtin_amdgcn_wmma_f32_16x16x32_bf16(
                    false, a_lo, false, b_hi, (short)0, acc[rt], false, false);
                acc[rt] = __builtin_amdgcn_wmma_f32_16x16x32_bf16(
                    false, a_hi, false, b_lo, (short)0, acc[rt], false, false);
            }
        }
    }

    // C/D layout: element r -> M = r + half*8, N = l16
    #pragma unroll
    for (int rt = 0; rt < 4; ++rt) {
        #pragma unroll
        for (int r = 0; r < 8; ++r) {
            int row = t0 + rt * 16 + half * 8 + r;
            float logit = acc[rt][r];
            scores[(size_t)row * EE + n0 + l16] = 1.f / (1.f + __expf(-logit));
        }
    }
}

// ---------------------------------------------------------------------------
// Kernel 2: per-token grouped top-k routing (one thread per token).
// Tie-break matches jax.lax.top_k (lowest index wins) via strict '>' scans.
// ---------------------------------------------------------------------------
__global__ __launch_bounds__(256) void gate_topk(
    const float* __restrict__ scores, const float* __restrict__ bias,
    float* __restrict__ out_idx, float* __restrict__ out_w,
    float* __restrict__ rowsum, int* __restrict__ cnt_be, int* __restrict__ cnt_e)
{
    int t = blockIdx.x * blockDim.x + threadIdx.x;
    if (t >= TT) return;
    const float* s = scores + (size_t)t * EE;

    float g1[NGROUP], g2[NGROUP];
    #pragma unroll
    for (int g = 0; g < NGROUP; ++g) { g1[g] = -1e30f; g2[g] = -1e30f; }
    float rsum = 0.f;
    for (int e = 0; e < EE; ++e) {
        float v   = s[e];
        rsum     += v;
        float bsc = v + bias[e];
        int   g   = e >> 5;
        if (bsc > g1[g])      { g2[g] = g1[g]; g1[g] = bsc; }
        else if (bsc > g2[g]) { g2[g] = bsc; }
    }
    rowsum[t] = rsum;

    // top-4 groups by (top1+top2) group score
    unsigned sel = 0;
    #pragma unroll
    for (int it = 0; it < TOPKG; ++it) {
        float best = -1e30f; int bg = 0;
        #pragma unroll
        for (int g = 0; g < NGROUP; ++g) {
            if (!((sel >> g) & 1u)) {
                float gs = g1[g] + g2[g];
                if (gs > best) { best = gs; bg = g; }
            }
        }
        sel |= 1u << bg;
    }

    // top-8 experts within selected groups; weights from UNbiased scores
    unsigned long long ch[4] = {0ull, 0ull, 0ull, 0ull};
    int   idx[TOPK];
    float wgt[TOPK];
    float denom = 0.f;
    for (int it = 0; it < TOPK; ++it) {
        float best = -1e30f; int be = 0;
        for (int e = 0; e < EE; ++e) {
            if (((sel >> (e >> 5)) & 1u) && !((ch[e >> 6] >> (e & 63)) & 1ull)) {
                float v = s[e] + bias[e];
                if (v > best) { best = v; be = e; }
            }
        }
        ch[be >> 6] |= 1ull << (be & 63);
        idx[it] = be;
        float wu = s[be];
        wgt[it] = wu;
        denom  += wu;
    }
    denom += 1e-20f;

    const int b = t / SEQQ;
    #pragma unroll
    for (int it = 0; it < TOPK; ++it) {
        out_idx[(size_t)t * TOPK + it] = (float)idx[it];
        out_w  [(size_t)t * TOPK + it] = wgt[it] / denom * SCALEF;
        atomicAdd(&cnt_be[b * EE + idx[it]], 1);
        atomicAdd(&cnt_e[idx[it]], 1);
    }
}

// ---------------------------------------------------------------------------
// Kernel 3: aux loss  = mean_b sum_e ce[b,e] * pmean[b,e] * ALPHA
// grid = BSZ blocks, 256 threads (one per expert); coalesced column reads.
// ---------------------------------------------------------------------------
__global__ __launch_bounds__(256) void gate_aux(
    const float* __restrict__ scores, const float* __restrict__ rowsum,
    const int* __restrict__ cnt_be, const int* __restrict__ cnt_e,
    float* __restrict__ out_loss, float* __restrict__ out_cnt)
{
    __shared__ float red[256];
    const int b = blockIdx.x;
    const int e = threadIdx.x;
    float acc = 0.f;
    for (int tt = 0; tt < SEQQ; ++tt) {
        int tok = b * SEQQ + tt;
        acc += scores[(size_t)tok * EE + e] / rowsum[tok];
    }
    float pmean = acc / (float)SEQQ;
    float ce = (float)cnt_be[b * EE + e] * ((float)EE / (float)(SEQQ * TOPK));
    red[e] = ce * pmean;
    __syncthreads();
    for (int off = 128; off > 0; off >>= 1) {
        if (e < off) red[e] += red[e + off];
        __syncthreads();
    }
    if (e == 0) atomicAdd(out_loss, red[0] * (ALPHAF / (float)BSZn));
    if (b == 0) out_cnt[e] = (float)cnt_e[e];
}

// ---------------------------------------------------------------------------
// Init: zero atomic counters and the loss slot (buffers are poisoned 0xAA)
// ---------------------------------------------------------------------------
__global__ void gate_init(int* cnt_be, int* cnt_e, float* out_loss)
{
    int i = blockIdx.x * blockDim.x + threadIdx.x;
    if (i < BSZn * EE) cnt_be[i] = 0;
    if (i < EE)        cnt_e[i]  = 0;
    if (i == 0)        *out_loss = 0.f;
}

extern "C" void kernel_launch(void* const* d_in, const int* in_sizes, int n_in,
                              void* d_out, int out_size, void* d_ws, size_t ws_size,
                              hipStream_t stream)
{
    (void)in_sizes; (void)n_in; (void)out_size; (void)ws_size;
    const float* x    = (const float*)d_in[0];   // [4,4096,2048] f32
    const float* w    = (const float*)d_in[1];   // [256,2048]    f32
    const float* bias = (const float*)d_in[2];   // [256]         f32
    float* out = (float*)d_out;

    // workspace layout
    float* scores = (float*)d_ws;                          // T*E f32 (16 MB)
    float* rowsum = scores + (size_t)TT * EE;              // T f32
    int*   cnt_be = (int*)(rowsum + TT);                   // BSZ*E int
    int*   cnt_e  = cnt_be + BSZn * EE;                    // E int

    // output layout: [topk_idx T*8][topk_weight T*8][aux_loss 1][counts E]
    float* out_idx  = out;
    float* out_w    = out + (size_t)TT * TOPK;
    float* out_loss = out + (size_t)2 * TT * TOPK;
    float* out_cnt  = out_loss + 1;

    gate_init<<<(BSZn * EE + 255) / 256, 256, 0, stream>>>(cnt_be, cnt_e, out_loss);
    gate_gemm_sigmoid<<<TT / ROWS, 512, 0, stream>>>(x, w, scores);
    gate_topk<<<TT / 256, 256, 0, stream>>>(scores, bias, out_idx, out_w,
                                            rowsum, cnt_be, cnt_e);
    gate_aux<<<BSZn, 256, 0, stream>>>(scores, rowsum, cnt_be, cnt_e,
                                       out_loss, out_cnt);
}
